// STGCNBlock_74431783240171
// MI455X (gfx1250) — matrix operand order repaired and checked by reading, not compiled, not run on hardware
//
#include <hip/hip_runtime.h>
#include <hip/hip_bf16.h>

// ---------------------------------------------------------------------------
// STGCN block for MI455X (gfx1250, wave32, WMMA, 320KB WGP LDS).
//   K1: h = relu(tconv1(x)) ; xw = h @ gcn_w   (bf16 WMMA, f32 acc, LDS-staged)
//   K2: per-graph GCN aggregation in 256KB LDS (ds_add_f32), bf16 output
//   K3: out = tconv2(gcn_out)                  (bf16 WMMA, async LDS staging)
//
// Reduction ordering for both temporal convs is tap-major: j' = dt*32 + c.
// Each WMMA K-step is one tap plane, so B-operand LDS planes are raw
// [node][channel] slabs -- contiguous copies, no im2col interleave.
// ---------------------------------------------------------------------------

#define Bn   4
#define CIN  32
#define COUT 32
#define Tn   48
#define Nn   2000
#define En   16000
#define BT   (Bn * Tn)          // 192 graphs
#define NCHUNK 16               // 128-node chunks per graph (last one clamped)

typedef __attribute__((ext_vector_type(16))) __bf16 v16bf;
typedef __attribute__((ext_vector_type(8)))  __bf16 v8bf;
typedef __attribute__((ext_vector_type(8)))  float  v8f;
typedef int v4i __attribute__((vector_size(16)));   // builtin's b128 unit

#ifndef __has_builtin
#define __has_builtin(x) 0
#endif
#if __has_builtin(__builtin_amdgcn_global_load_async_to_lds_b128)
#define ASYNC_LDS 1
typedef __attribute__((address_space(1))) v4i* gv4i_p;   // global int4*
typedef __attribute__((address_space(3))) v4i* lv4i_p;   // LDS int4*
#else
#define ASYNC_LDS 0
#endif

// ---- WMMA fragment layouts (CDNA5 ISA 7.12.2, wave32) ---------------------
// A (16x32 bf16): M = lane&15 ; K(e) = e + (e>=8 ? 8 : 0) + (lane>=16 ? 8 : 0)
//   -> elements 0..7 K-contiguous at +hi*8, elements 8..15 at +16.
// B (32x16 bf16): N = lane&15 ; K(e) same as A -> 0..7 at +hi*8, 8..15 at +16.
// C/D (16x16 f32): N = lane&15 ; M(r) = r + (lane>=16 ? 8 : 0)
__device__ __forceinline__ v16bf ld_frag(const __bf16* p, int second_half_off) {
    v8bf lo = *(const v8bf*)p;
    v8bf hi = *(const v8bf*)(p + second_half_off);
    return __builtin_shufflevector(lo, hi, 0, 1, 2, 3, 4, 5, 6, 7,
                                           8, 9, 10, 11, 12, 13, 14, 15);
}
#define WMMA_BF16(A, B, C) \
    __builtin_amdgcn_wmma_f32_16x16x32_bf16(false, (A), false, (B), (short)0, (C), false, false)

// ---------------------------------------------------------------------------
// K0a: deg[n] = 1.0  (self loop)
__global__ __launch_bounds__(256) void k_deg_init(float* __restrict__ deg) {
    int i = blockIdx.x * blockDim.x + threadIdx.x;
    if (i < Nn) deg[i] = 1.0f;
}

// K0b: deg[dst] += 1 per edge
__global__ __launch_bounds__(256) void k_deg_count(const long long* __restrict__ ei,
                                                   float* __restrict__ deg) {
    int e = blockIdx.x * blockDim.x + threadIdx.x;
    if (e < En) atomicAdd(&deg[(int)ei[En + e]], 1.0f);
}

// ---------------------------------------------------------------------------
// K1: block = 8 waves, one (b,t) graph x 128-node chunk.
//   GEMM1: A1 = w1 permuted [32 x 96] (j' = dt*32+ci), B1 = tap planes of x
//   bias+ReLU staged to LDS (acc layout -> K-major B layout), then
//   GEMM2: A2 = gcn_w^T [32 x 32], B2 = h.
__global__ __launch_bounds__(256) void k1_tconv_gcnw(const float* __restrict__ x,
                                                     const float* __restrict__ w1,
                                                     const float* __restrict__ b1,
                                                     const float* __restrict__ gw,
                                                     float* __restrict__ xw) {
    __shared__ __bf16 sW[32 * 96];        // w1 tap-major, row m          (6 KB)
    __shared__ __bf16 sG[32 * 32];        // gcn_w^T, row co2, K-major    (2 KB)
    __shared__ __bf16 sX[3 * 128 * 32];   // tap planes [dt][node][ci]   (24 KB)
    __shared__ __bf16 sH[128 * 32];       // relu(h) [node][c]            (8 KB)

    const int tid   = threadIdx.x;
    const int g     = blockIdx.x >> 4;      // graph = b*T + t
    const int chunk = blockIdx.x & (NCHUNK - 1);
    const int b     = g / Tn;
    const int t     = g % Tn;
    int n_base = chunk * 128;
    if (n_base > Nn - 128) n_base = Nn - 128;   // clamp: duplicate, identical work

    // ---- stage weights (tap-major permutation j' = dt*32 + ci) ------------
    for (int i = tid; i < 32 * 32; i += 256) {
        const int mm = i >> 5, ci = i & 31;
        #pragma unroll
        for (int dt = 0; dt < 3; ++dt)
            sW[mm * 96 + dt * 32 + ci] = (__bf16)w1[mm * 96 + ci * 3 + dt];
    }
    for (int i = tid; i < 32 * 32; i += 256) {
        const int c = i >> 5, co = i & 31;
        sG[co * 32 + c] = (__bf16)gw[i];        // transpose: A2[co2][c]
    }

    // ---- stage x tap planes: sX[dt][nl][ci], coalesced b128 loads ---------
    #pragma unroll
    for (int dt = 0; dt < 3; ++dt) {
        const int  tt    = t + dt - 1;
        const bool valid = (tt >= 0) && (tt < Tn);     // uniform branch
        __bf16* plane = &sX[dt * 128 * 32];
        #pragma unroll
        for (int p = 0; p < 4; ++p) {
            const int ci  = p * 8 + (tid >> 5);
            const int nl4 = (tid & 31) * 4;
            float4 v = make_float4(0.f, 0.f, 0.f, 0.f);
            if (valid)
                v = *(const float4*)&x[(((size_t)b * CIN + ci) * Tn + tt) * Nn
                                       + n_base + nl4];
            plane[(nl4 + 0) * 32 + ci] = (__bf16)v.x;
            plane[(nl4 + 1) * 32 + ci] = (__bf16)v.y;
            plane[(nl4 + 2) * 32 + ci] = (__bf16)v.z;
            plane[(nl4 + 3) * 32 + ci] = (__bf16)v.w;
        }
    }
    __syncthreads();

    const int  lane = tid & 31;
    const int  wave = tid >> 5;
    const bool hi   = lane >= 16;
    const int  col  = lane & 15;
    const int  m    = col;                   // A-fragment row held by this lane
    const int  nl   = wave * 16 + col;       // node within 128-chunk

    // ---- GEMM1: 3 K-steps, one per tap ------------------------------------
    v8f c0 = {}, c1 = {};
    #pragma unroll
    for (int s = 0; s < 3; ++s) {
        const v16bf a0 = ld_frag(&sW[m * 96 + s * 32 + (hi ? 8 : 0)], 16);
        const v16bf a1 = ld_frag(&sW[(m + 16) * 96 + s * 32 + (hi ? 8 : 0)], 16);
        const v16bf bb = ld_frag(&sX[s * 128 * 32 + nl * 32 + (hi ? 8 : 0)], 16);
        c0 = WMMA_BF16(a0, bb, c0);
        c1 = WMMA_BF16(a1, bb, c1);
    }

    // ---- bias + ReLU -> sH[nl][c] (K-major for GEMM2's B operand) ---------
    {
        __bf16* hb = &sH[nl * 32];
        #pragma unroll
        for (int r = 0; r < 8; ++r) {
            const int m0 = r + (hi ? 8 : 0);
            hb[m0]      = (__bf16)fmaxf(c0[r] + b1[m0], 0.f);
            hb[m0 + 16] = (__bf16)fmaxf(c1[r] + b1[m0 + 16], 0.f);
        }
    }
    __syncthreads();

    // ---- GEMM2: xw[n, co2] = sum_c h[n, c] * gcn_w[c, co2] ----------------
    const v16bf a20 = ld_frag(&sG[m * 32 + (hi ? 8 : 0)], 16);
    const v16bf a21 = ld_frag(&sG[(m + 16) * 32 + (hi ? 8 : 0)], 16);
    const v16bf b2f = ld_frag(&sH[nl * 32 + (hi ? 8 : 0)], 16);
    v8f d0 = {}, d1 = {};
    d0 = WMMA_BF16(a20, b2f, d0);
    d1 = WMMA_BF16(a21, b2f, d1);

    // ---- store xw as [g][n][c]: float2 (b64) stores -----------------------
    const size_t base = ((size_t)g * Nn + n_base + nl) * 32;
    #pragma unroll
    for (int r = 0; r < 8; r += 2) {
        const int m0 = r + (hi ? 8 : 0);
        float2 lo2 = make_float2(d0[r], d0[r + 1]);
        float2 hi2 = make_float2(d1[r], d1[r + 1]);
        *(float2*)&xw[base + m0]      = lo2;
        *(float2*)&xw[base + m0 + 16] = hi2;
    }
}

// ---------------------------------------------------------------------------
// K2: one workgroup owns one (b,t) graph. agg lives entirely in LDS
// (2000*32*4B = 256KB <= 320KB WGP LDS). One lane per channel: the 32-float
// xw row gather is coalesced and ds_add_f32 atomics hit 32 distinct banks.
// Output hg is bf16 [g][n][c] (feeds a bf16 WMMA; halves HBM traffic).
extern __shared__ float s_agg[];
__global__ __launch_bounds__(256) void k2_aggregate(const float* __restrict__ xw,
                                                    const long long* __restrict__ ei,
                                                    const float* __restrict__ deg,
                                                    const float* __restrict__ gb,
                                                    __bf16* __restrict__ hg) {
    const int g = blockIdx.x;
    const long long* srcs = ei;
    const long long* dsts = ei + En;
    const float* xg = xw + (size_t)g * Nn * 32;

    for (int i = threadIdx.x; i < Nn * 32; i += blockDim.x) s_agg[i] = 0.0f;
    __syncthreads();

    const int lane   = threadIdx.x & 31;
    const int wave   = threadIdx.x >> 5;
    const int nwaves = blockDim.x >> 5;

    for (int e = wave; e < En; e += nwaves) {
        __builtin_prefetch(&srcs[e + 4 * nwaves], 0, 0);   // global_prefetch_b8
        __builtin_prefetch(&dsts[e + 4 * nwaves], 0, 0);
        const int s = (int)srcs[e];
        const int d = (int)dsts[e];
        const float w = __frsqrt_rn(deg[s]) * __frsqrt_rn(deg[d]);
        const float v = xg[(size_t)s * 32 + lane] * w;     // coalesced row gather
        atomicAdd(&s_agg[d * 32 + lane], v);               // ds_add_f32
    }
    __syncthreads();

    // out = agg + (1/deg) * xw + bias, stored bf16 as [g][n][c]
    for (int i = threadIdx.x; i < Nn * 32; i += blockDim.x) {
        const int n = i >> 5;
        const int c = i & 31;
        const float di = __frsqrt_rn(deg[n]);
        hg[(size_t)g * Nn * 32 + i] =
            (__bf16)(s_agg[i] + (di * di) * xg[i] + gb[c]);
    }
}

// ---------------------------------------------------------------------------
// K3: tconv2 over hg(bf16) [BT][N][C] -> out [B][C][T][N] (NCHW).
// Tap planes are contiguous 8KB slabs of hg -> async global->LDS copies.
__global__ __launch_bounds__(256) void k3_tconv2(const __bf16* __restrict__ hg,
                                                 const float* __restrict__ w2,
                                                 const float* __restrict__ b2,
                                                 float* __restrict__ out) {
    __shared__ __bf16 sW[32 * 96];        // w2 tap-major                 (6 KB)
    __shared__ __bf16 sX[3 * 128 * 32];   // tap planes [dt][node][c]    (24 KB)

    const int tid   = threadIdx.x;
    const int g     = blockIdx.x >> 4;
    const int chunk = blockIdx.x & (NCHUNK - 1);
    const int b     = g / Tn;
    const int t     = g % Tn;
    int n_base = chunk * 128;
    if (n_base > Nn - 128) n_base = Nn - 128;

    // weights, tap-major permutation j' = dt*32 + c
    for (int i = tid; i < 32 * 32; i += 256) {
        const int mm = i >> 5, c = i & 31;
        #pragma unroll
        for (int dt = 0; dt < 3; ++dt)
            sW[mm * 96 + dt * 32 + c] = (__bf16)w2[mm * 96 + c * 3 + dt];
    }

    // ---- stage hg tap planes: contiguous 8KB copies ------------------------
    #pragma unroll
    for (int dt = 0; dt < 3; ++dt) {
        const int tt = t + dt - 1;
        __bf16* plane = &sX[dt * 128 * 32];
        if (tt >= 0 && tt < Tn) {                          // uniform branch
            const __bf16* src = hg + (((size_t)b * Tn + tt) * Nn + n_base) * 32;
#if ASYNC_LDS
            #pragma unroll
            for (int p = 0; p < 2; ++p) {
                const int off = (p * 256 + tid) * 8;       // 16B per lane
                __builtin_amdgcn_global_load_async_to_lds_b128(
                    (gv4i_p)(src + off),                   // global int4*
                    (lv4i_p)(plane + off),                 // LDS int4*
                    0, 0);
            }
#else
            #pragma unroll
            for (int p = 0; p < 2; ++p) {
                const int off = (p * 256 + tid) * 8;
                *(v8bf*)(plane + off) = *(const v8bf*)(src + off);
            }
#endif
        } else {
            const v8bf z = {};
            for (int i = tid; i < 512; i += 256) *(v8bf*)(plane + i * 8) = z;
        }
    }
#if ASYNC_LDS
#if __has_builtin(__builtin_amdgcn_s_wait_asynccnt)
    __builtin_amdgcn_s_wait_asynccnt(0);
#else
    asm volatile("s_wait_asynccnt 0x0" ::: "memory");
#endif
#endif
    __syncthreads();

    const int  lane = tid & 31;
    const int  wave = tid >> 5;
    const bool hi   = lane >= 16;
    const int  col  = lane & 15;
    const int  m    = col;
    const int  nl   = wave * 16 + col;

    v8f c0 = {}, c1 = {};
    #pragma unroll
    for (int s = 0; s < 3; ++s) {
        const v16bf a0 = ld_frag(&sW[m * 96 + s * 32 + (hi ? 8 : 0)], 16);
        const v16bf a1 = ld_frag(&sW[(m + 16) * 96 + s * 32 + (hi ? 8 : 0)], 16);
        const v16bf bb = ld_frag(&sX[s * 128 * 32 + nl * 32 + (hi ? 8 : 0)], 16);
        c0 = WMMA_BF16(a0, bb, c0);
        c1 = WMMA_BF16(a1, bb, c1);
    }

    const int n = n_base + nl;
    #pragma unroll
    for (int r = 0; r < 8; ++r) {
        const int m0 = r + (hi ? 8 : 0);
        out[(((size_t)b * COUT + m0) * Tn + t) * Nn + n]      = c0[r] + b2[m0];
        out[(((size_t)b * COUT + m0 + 16) * Tn + t) * Nn + n] = c1[r] + b2[m0 + 16];
    }
}

// ---------------------------------------------------------------------------
extern "C" void kernel_launch(void* const* d_in, const int* in_sizes, int n_in,
                              void* d_out, int out_size, void* d_ws, size_t ws_size,
                              hipStream_t stream) {
    const float*     x   = (const float*)d_in[0];
    const long long* ei  = (const long long*)d_in[1];   // int64 edge_index [2,E]
    const float*     w1  = (const float*)d_in[2];
    const float*     b1  = (const float*)d_in[3];
    const float*     gw  = (const float*)d_in[4];
    const float*     gb  = (const float*)d_in[5];
    const float*     w2  = (const float*)d_in[6];
    const float*     b2  = (const float*)d_in[7];
    float*           out = (float*)d_out;

    // workspace layout: xw (f32) | hg (bf16) | deg (f32)
    const size_t tensor_elems = (size_t)BT * Nn * 32;   // 12.288M elements
    float*  xw  = (float*)d_ws;
    __bf16* hgb = (__bf16*)(xw + tensor_elems);
    float*  deg = (float*)(hgb + tensor_elems);

    k_deg_init <<<(Nn + 255) / 256, 256, 0, stream>>>(deg);
    k_deg_count<<<(En + 255) / 256, 256, 0, stream>>>(ei, deg);

    const int blocks = BT * NCHUNK;    // 192 * 16 = 3072 blocks of 8 waves
    k1_tconv_gcnw<<<blocks, 256, 0, stream>>>(x, w1, b1, gw, xw);

    const size_t lds_bytes = (size_t)Nn * 32 * sizeof(float);   // 256 KB / WGP
    k2_aggregate<<<BT, 256, lds_bytes, stream>>>(xw, ei, deg, gb, hgb);

    k3_tconv2<<<blocks, 256, 0, stream>>>(hgb, w2, b2, out);
}
